// AnyTSRpp_64596308132305
// MI455X (gfx1250) — matrix operand order repaired and checked by reading, not compiled, not run on hardware
//
#include <hip/hip_runtime.h>
#include <hip/hip_bf16.h>
#include <math.h>

typedef __attribute__((ext_vector_type(16))) _Float16 v16h;
typedef __attribute__((ext_vector_type(8)))  float    v8f;

#define CF    64
#define HQ    256
#define WQ    256
#define NPOS  (HQ*WQ)   /* 65536 */
#define BATCH 2

/* workspace byte offsets (all 32B aligned) */
#define WS_ATTN   0                            /* f32 [8][64][64] accumulator  */
#define WS_ATTNF  (131072)                     /* f16 frag [8][2][4][32][16]   */
#define WS_W00F   (WS_ATTNF + 8*4096*2)        /* f16 frag [17][16][32][16]    */
#define WS_W1F    (WS_W00F + 17*16*512*2)      /* f16 frag [8][16][32][16]     */
#define WS_WVF    (WS_W1F  + 8*16*512*2)       /* f16 frag [2][4][32][16]      */
#define WS_WKF    (WS_WVF  + 2*4*512*2)        /* f16 frag [2][4][32][16]      */

__device__ __forceinline__ v8f zero8() {
  v8f z;
#pragma unroll
  for (int t = 0; t < 8; ++t) z[t] = 0.f;
  return z;
}

__device__ __forceinline__ v8f wmma16(v16h a, v16h b, v8f c) {
  /* D = A(16x32 f16) * B(32x16 f16) + C(16x16 f32) */
  return __builtin_amdgcn_wmma_f32_16x16x32_f16(false, a, false, b, (short)0, c,
                                                false, false);
}

/* contiguous 32B fragment load (weights pre-swizzled to fragment order) */
__device__ __forceinline__ v16h load_frag(const _Float16* p) {
  return *(const v16h*)p;
}

/* A-matrix 16x32 f16 fragment gather from plain storage:
   element(m,k) = src[m*sM + k*sK].
   Layout: lanes 0-15 hold m=L, K in {0..15}; lanes 16-31 K in {16..31}. */
__device__ __forceinline__ v16h gather_a(const _Float16* src, int sM, int sK) {
  const int lane = threadIdx.x & 31;
  const int m = lane & 15, hb = (lane >> 4) & 1;
  v16h a;
#pragma unroll
  for (int t = 0; t < 16; ++t) {
    const int kk = t + hb * 16;
    a[t] = src[m * sM + kk * sK];
  }
  return a;
}

/* B-matrix 32x16 f16 fragment gather: element(k,n) = src[k*sK + n*sN].
   Lane holds n=L&15; halves t -> K = t + 16*(lane>=16). */
__device__ __forceinline__ v16h gather_b_h(const _Float16* src, int sK, int sN) {
  const int lane = threadIdx.x & 31;
  const int n = lane & 15, hb = (lane >> 4) & 1;
  v16h b;
#pragma unroll
  for (int t = 0; t < 16; ++t) {
    const int k = t + 16 * hb;
    b[t] = src[k * sK + n * sN];
  }
  return b;
}

/* fragment-order half index decode helpers used by prep/softmax:
   within one [kb][nb][32][16] frag array: n = nb*16+(lane&15), k = kb*32 + t + 16*(lane>>4) */

/* ------------------------------------------------------------------ */
/* Kernel 0: zero attention accumulator, swizzle weights to f16 frags */
__global__ void prep_kernel(const float* __restrict__ Wk,
                            const float* __restrict__ Wv,
                            const float* __restrict__ W00,
                            const float* __restrict__ W1,
                            float* attnWS, _Float16* W00F, _Float16* W1F,
                            _Float16* WvF, _Float16* WkF) {
  const int nW00 = 17 * 16 * 512;   /* halves */
  const int nW1  = 8 * 16 * 512;
  const int nWv  = 2 * 4 * 512;
  const int total = 32768 + nW00 + nW1 + nWv + nWv;
  for (int i = blockIdx.x * blockDim.x + threadIdx.x; i < total;
       i += gridDim.x * blockDim.x) {
    int r = i;
    if (r < 32768) { attnWS[r] = 0.f; continue; }
    r -= 32768;
    if (r < nW00) {
      const int t = r & 15, lane = (r >> 4) & 31, nb = (r >> 9) & 15, kb = r >> 13;
      const int n = nb * 16 + (lane & 15);
      const int k = kb * 32 + t + 16 * (lane >> 4);
      W00F[r] = (k < 516) ? (_Float16)W00[n * 516 + k] : (_Float16)0.f;
      continue;
    }
    r -= nW00;
    if (r < nW1) {
      const int t = r & 15, lane = (r >> 4) & 31, nb = (r >> 9) & 15, kb = r >> 13;
      const int n = nb * 16 + (lane & 15);
      const int k = kb * 32 + t + 16 * (lane >> 4);
      W1F[r] = (_Float16)W1[n * 256 + k];
      continue;
    }
    r -= nW1;
    if (r < nWv) {
      const int t = r & 15, lane = (r >> 4) & 31, nb = (r >> 9) & 3, kb = r >> 11;
      const int n = nb * 16 + (lane & 15);
      const int k = kb * 32 + t + 16 * (lane >> 4);
      WvF[r] = (_Float16)Wv[n * 64 + k];
      continue;
    }
    r -= nWv;
    {
      const int t = r & 15, lane = (r >> 4) & 31, nb = (r >> 9) & 3, kb = r >> 11;
      const int n = nb * 16 + (lane & 15);
      const int k = kb * 32 + t + 16 * (lane >> 4);
      WkF[r] = (_Float16)Wk[n * 64 + k];
    }
  }
}

/* ------------------------------------------------------------------ */
/* Kernel 1: accumulate raw attention logits A[b][o][q][k] = sum_n q*k */
#define K1_GRIDX 128
__global__ __launch_bounds__(128) void attn_accum_kernel(
    const float* __restrict__ feat, const float* __restrict__ coord,
    const float* __restrict__ Wq, const float* __restrict__ bq,
    const float* __restrict__ bk, const float* __restrict__ ls,
    const _Float16* __restrict__ WkF, float* attnWS) {
  __shared__ _Float16 fsL[32 * 64];   /* [pos][ch] */
  __shared__ _Float16 qT[64 * 32];    /* [qch][pos] */
  __shared__ _Float16 kT[64 * 32];    /* [kch][pos] */
  __shared__ int   iyA[32], ixA[32];
  __shared__ float wtA[32], r0A[32], r1A[32];

  const int tid = threadIdx.x;
  const int wave = tid >> 5, lane = tid & 31;
  const int bo = blockIdx.y;
  const int b = bo >> 2, o = bo & 3;
  const float vx = (o < 2) ? -1.f : 1.f;
  const float vy = (o & 1) ? 1.f : -1.f;
  const float inv2ls2 = -0.5f / (ls[0] * ls[0]);

  v8f acc[4];
#pragma unroll
  for (int t = 0; t < 4; ++t) acc[t] = zero8();

  for (int ch = 0; ch < NPOS / (K1_GRIDX * 32); ++ch) {
    const int pbase = blockIdx.x * (NPOS / K1_GRIDX) + ch * 32;
    if (tid < 32) {
      const int p = pbase + tid;
      const int ii = p >> 8, jj = p & 255;
      const float cy0 = coord[((b * HQ + ii) * WQ + jj) * 2 + 0];
      const float cx0 = coord[((b * HQ + ii) * WQ + jj) * 2 + 1];
      const float cy = fminf(fmaxf(cy0 + vx * (1.f / 64.f) + 1e-6f, -1.f + 1e-6f), 1.f - 1e-6f);
      const float cx = fminf(fmaxf(cx0 + vy * (1.f / 64.f) + 1e-6f, -1.f + 1e-6f), 1.f - 1e-6f);
      const int iy = (int)fminf(fmaxf(roundf((cy + 1.f) * 32.f - 0.5f), 0.f), 63.f);
      const int ix = (int)fminf(fmaxf(roundf((cx + 1.f) * 32.f - 0.5f), 0.f), 63.f);
      const float oy = ((float)iy + 0.5f) * (2.f / 64.f) - 1.f;
      const float ox = ((float)ix + 0.5f) * (2.f / 64.f) - 1.f;
      const float r0 = cy0 - oy, r1 = cx0 - ox;
      const float rd = (r0 * 64.f) * (r0 * 64.f) + (r1 * 64.f) * (r1 * 64.f);
      iyA[tid] = iy; ixA[tid] = ix;
      wtA[tid] = expf(rd * inv2ls2);
      r0A[tid] = r0; r1A[tid] = r1;
    }
    __syncthreads();
    { /* fs + q generation: (pos = tid&31, chans = (tid>>5)*16..+16) */
      const int p = tid & 31;
      const int c0 = (tid >> 5) * 16;
      const int iy = iyA[p], ix = ixA[p];
      const float wt = wtA[p], r0 = r0A[p], r1 = r1A[p];
      const float* fp = feat + b * CF * 4096 + iy * 64 + ix;
#pragma unroll
      for (int jj = 0; jj < 16; ++jj) {
        const int c = c0 + jj;
        fsL[p * 64 + c] = (_Float16)(fp[c * 4096] * wt);
        const float qv = Wq[c * 2 + 0] * r0 + Wq[c * 2 + 1] * r1 + bq[c];
        qT[c * 32 + p] = (_Float16)fmaxf(qv, 0.f);
      }
    }
    __syncthreads();
    /* k = relu(fs @ Wk^T + bk): 8 tiles over (2 posblk x 4 chblk) */
#pragma unroll
    for (int tsel = 0; tsel < 2; ++tsel) {
      const int tile = wave + tsel * 4;
      const int mb = tile >> 2, nb = tile & 3;
      v8f c = zero8();
#pragma unroll
      for (int kb = 0; kb < 2; ++kb) {
        const v16h a  = gather_a(fsL + mb * 16 * 64 + kb * 32, 64, 1);
        const v16h bb = load_frag(WkF + ((kb * 4 + nb) * 32 + lane) * 16);
        c = wmma16(a, bb, c);
      }
      const int n = lane & 15, hb = lane >> 4;
      const float bkn = bk[nb * 16 + n];
#pragma unroll
      for (int r = 0; r < 8; ++r) {
        const int m = r + 8 * hb;
        kT[(nb * 16 + n) * 32 + mb * 16 + m] = (_Float16)fmaxf(c[r] + bkn, 0.f);
      }
    }
    __syncthreads();
    /* A += q^T x k over K=32 positions: wave owns q-block = wave */
#pragma unroll
    for (int nb = 0; nb < 4; ++nb) {
      const v16h a  = gather_a(qT + wave * 16 * 32, 32, 1);  /* (m=qch, k=pos) */
      const v16h bb = gather_b_h(kT + nb * 16 * 32, 1, 32);  /* (k=pos, n=kch) */
      acc[nb] = wmma16(a, bb, acc[nb]);
    }
    __syncthreads();
  }
  { /* flush to global via f32 atomics */
    const int n = lane & 15, hb = lane >> 4;
    float* ap = attnWS + bo * 4096;
#pragma unroll
    for (int nb = 0; nb < 4; ++nb)
#pragma unroll
      for (int r = 0; r < 8; ++r) {
        const int qi = wave * 16 + r + 8 * hb;
        atomicAdd(&ap[qi * 64 + nb * 16 + n], acc[nb][r]);
      }
  }
}

/* ------------------------------------------------------------------ */
/* Kernel 2: softmax rows of A[bo][64][64], emit f16 B-fragments      */
__global__ __launch_bounds__(64) void softmax_kernel(const float* __restrict__ attnWS,
                                                     _Float16* __restrict__ attnF) {
  __shared__ float sm[64 * 64];
  const int bo = blockIdx.x, tid = threadIdx.x;
  {
    const float* row = attnWS + (bo * 64 + tid) * 64;
    float mx = -1e30f;
    for (int k = 0; k < 64; ++k) mx = fmaxf(mx, row[k]);
    float s = 0.f;
    for (int k = 0; k < 64; ++k) s += expf(row[k] - mx);
    const float inv = 1.f / s;
    for (int k = 0; k < 64; ++k) sm[tid * 64 + k] = expf(row[k] - mx) * inv;
  }
  __syncthreads();
  /* fragment order: B(k,n=q) = attn[q][k]; idx = ((kb*4+nb)*32+lane)*16 + t */
  for (int idx = tid; idx < 4096; idx += 64) {
    const int t = idx & 15, lane = (idx >> 4) & 31, nb = (idx >> 9) & 3, kb = idx >> 11;
    const int q = nb * 16 + (lane & 15);
    const int k = kb * 32 + t + 16 * (lane >> 4);
    attnF[bo * 4096 + idx] = (_Float16)sm[q * 64 + k];
  }
}

/* ------------------------------------------------------------------ */
/* Kernel 3: fused per-position pipeline, one wave = 16 positions     */
__global__ __launch_bounds__(64) void fused_kernel(
    const float* __restrict__ feat, const float* __restrict__ inp,
    const float* __restrict__ coord, const float* __restrict__ cell,
    const float* __restrict__ scale, const float* __restrict__ bv,
    const float* __restrict__ b00, const float* __restrict__ b1,
    const float* __restrict__ W2, const float* __restrict__ b2,
    const float* __restrict__ ls, const _Float16* __restrict__ attnF,
    const _Float16* __restrict__ W00F, const _Float16* __restrict__ W1F,
    const _Float16* __restrict__ WvF, float* __restrict__ out) {
  __shared__ _Float16 gridL[2][16 * 544]; /* grid activations, padded to 544 ch */
  __shared__ _Float16 x0L[2][16 * 256];
  __shared__ _Float16 vLs[2][16 * 64];
  __shared__ float bilinL[2][16];
  __shared__ float outacc[2][16];

  const int wave = threadIdx.x >> 5, lane = threadIdx.x & 31;
  const int m = lane & 15, hb = lane >> 4, n16 = lane & 15;
  const int pbase = (blockIdx.x * 2 + wave) * 16;
  const int b = pbase >> 16;
  const int pin = pbase & (NPOS - 1);
  const int p = pin + m;
  const int ii = p >> 8, jj = p & 255;

  _Float16* gL = gridL[wave];
  _Float16* x0 = x0L[wave];
  _Float16* vL = vLs[wave];

  const float cy0 = coord[((b * HQ + ii) * WQ + jj) * 2 + 0];
  const float cx0 = coord[((b * HQ + ii) * WQ + jj) * 2 + 1];
  const float inv2ls2 = -0.5f / (ls[0] * ls[0]);

  if (hb == 0) {
    outacc[wave][m] = 0.f;
    /* bilinear border sample of inp */
    const float y = fminf(fmaxf((cy0 + 1.f) * 32.f - 0.5f, 0.f), 63.f);
    const float x = fminf(fmaxf((cx0 + 1.f) * 32.f - 0.5f, 0.f), 63.f);
    const float y0 = floorf(y), x0f = floorf(x);
    const float wy = y - y0, wx = x - x0f;
    const int y0i = min(max((int)y0, 0), 63), y1i = min(max((int)y0 + 1, 0), 63);
    const int x0i = min(max((int)x0f, 0), 63), x1i = min(max((int)x0f + 1, 0), 63);
    const float* ip = inp + b * 4096;
    const float v00 = ip[y0i * 64 + x0i], v01 = ip[y0i * 64 + x1i];
    const float v10 = ip[y1i * 64 + x0i], v11 = ip[y1i * 64 + x1i];
    bilinL[wave][m] = v00 * (1.f - wy) * (1.f - wx) + v01 * (1.f - wy) * wx +
                      v10 * wy * (1.f - wx) + v11 * wy * wx;
    gL[m * 544 + 512] = (_Float16)(cell[b * 2 + 0] * 64.f);
    gL[m * 544 + 513] = (_Float16)(cell[b * 2 + 1] * 64.f);
    gL[m * 544 + 514] = (_Float16)(scale[b * 2 + 0]);
    gL[m * 544 + 515] = (_Float16)(scale[b * 2 + 1]);
#pragma unroll
    for (int c = 516; c < 544; ++c) gL[m * 544 + c] = (_Float16)0.f;
  }
  __syncthreads();

  for (int o = 0; o < 4; ++o) {
    const float vx = (o < 2) ? -1.f : 1.f;
    const float vy = (o & 1) ? 1.f : -1.f;
    const float cy = fminf(fmaxf(cy0 + vx * (1.f / 64.f) + 1e-6f, -1.f + 1e-6f), 1.f - 1e-6f);
    const float cx = fminf(fmaxf(cx0 + vy * (1.f / 64.f) + 1e-6f, -1.f + 1e-6f), 1.f - 1e-6f);
    const int iy = (int)fminf(fmaxf(roundf((cy + 1.f) * 32.f - 0.5f), 0.f), 63.f);
    const int ix = (int)fminf(fmaxf(roundf((cx + 1.f) * 32.f - 0.5f), 0.f), 63.f);
    const float oy = ((float)iy + 0.5f) * (2.f / 64.f) - 1.f;
    const float ox = ((float)ix + 0.5f) * (2.f / 64.f) - 1.f;
    const float r0 = cy0 - oy, r1 = cx0 - ox;
    const float wt = expf(((r0 * 64.f) * (r0 * 64.f) + (r1 * 64.f) * (r1 * 64.f)) * inv2ls2);

    /* fs generation: lane covers channels hb*32..+32 of its position m */
    const float* fp = feat + b * CF * 4096 + iy * 64 + ix;
#pragma unroll
    for (int cc = 0; cc < 32; ++cc) {
      const int c = hb * 32 + cc;
      gL[m * 544 + 256 + o * 64 + c] = (_Float16)(fp[c * 4096] * wt);
    }
    __syncthreads();

    /* v = relu(fs @ Wv^T + bv) -> vL */
    for (int nb = 0; nb < 4; ++nb) {
      v8f c = zero8();
#pragma unroll
      for (int kb = 0; kb < 2; ++kb) {
        const v16h a  = gather_a(gL + 256 + o * 64 + kb * 32, 544, 1);
        const v16h bb = load_frag(WvF + ((kb * 4 + nb) * 32 + lane) * 16);
        c = wmma16(a, bb, c);
      }
      const float bvn = bv[nb * 16 + n16];
#pragma unroll
      for (int r = 0; r < 8; ++r)
        vL[(r + 8 * hb) * 64 + nb * 16 + n16] = (_Float16)fmaxf(c[r] + bvn, 0.f);
    }
    __syncthreads();

    /* off = v @ attn^T -> grid channels o*64..+64 */
    const _Float16* af = attnF + (b * 4 + o) * 4096;
    for (int nb = 0; nb < 4; ++nb) {
      v8f c = zero8();
#pragma unroll
      for (int kb = 0; kb < 2; ++kb) {
        const v16h a  = gather_a(vL + kb * 32, 64, 1);
        const v16h bb = load_frag(af + ((kb * 4 + nb) * 32 + lane) * 16);
        c = wmma16(a, bb, c);
      }
#pragma unroll
      for (int r = 0; r < 8; ++r)
        gL[(r + 8 * hb) * 544 + o * 64 + nb * 16 + n16] = (_Float16)c[r];
    }
    __syncthreads();
  }

  /* x0 = grid @ W00^T + b00 (K = 544 padded, N = 256) */
  for (int nb = 0; nb < 16; ++nb) {
    v8f c = zero8();
    for (int kb = 0; kb < 17; ++kb) {
      const _Float16* wp = W00F + ((kb * 16 + nb) * 32 + lane) * 16;
      __builtin_prefetch(wp + 16 * 512, 0, 1);    /* next kb slice */
      const v16h a  = gather_a(gL + kb * 32, 544, 1);
      const v16h bb = load_frag(wp);
      c = wmma16(a, bb, c);
    }
    const float bn = b00[nb * 16 + n16];
#pragma unroll
    for (int r = 0; r < 8; ++r)
      x0[(r + 8 * hb) * 256 + nb * 16 + n16] = (_Float16)(c[r] + bn);
  }
  __syncthreads();

  /* out += W2 . gelu(x0 @ W1^T + b1), W2 folded on the fly */
  float par[8];
#pragma unroll
  for (int r = 0; r < 8; ++r) par[r] = 0.f;
  for (int nb = 0; nb < 16; ++nb) {
    v8f c = zero8();
    for (int kb = 0; kb < 8; ++kb) {
      const _Float16* wp = W1F + ((kb * 16 + nb) * 32 + lane) * 16;
      __builtin_prefetch(wp + 16 * 512, 0, 1);
      const v16h a  = gather_a(x0 + kb * 32, 256, 1);
      const v16h bb = load_frag(wp);
      c = wmma16(a, bb, c);
    }
    const int n = nb * 16 + n16;
    const float b1n = b1[n], w2n = W2[n];
#pragma unroll
    for (int r = 0; r < 8; ++r) {
      const float xv = c[r] + b1n;
      const float g = 0.5f * xv * (1.f + erff(xv * 0.70710678118f)); /* exact gelu */
      par[r] += g * w2n;
    }
  }
#pragma unroll
  for (int r = 0; r < 8; ++r) atomicAdd(&outacc[wave][r + 8 * hb], par[r]);
  __syncthreads();

  if (hb == 0) out[b * NPOS + pin + m] = outacc[wave][m] + b2[0] + bilinL[wave][m];
}

/* ------------------------------------------------------------------ */
extern "C" void kernel_launch(void* const* d_in, const int* in_sizes, int n_in,
                              void* d_out, int out_size, void* d_ws, size_t ws_size,
                              hipStream_t stream) {
  const float* feat  = (const float*)d_in[0];
  const float* inp   = (const float*)d_in[1];
  const float* coord = (const float*)d_in[2];
  const float* cell  = (const float*)d_in[3];
  const float* scale = (const float*)d_in[4];
  const float* Wq    = (const float*)d_in[5];
  const float* bq    = (const float*)d_in[6];
  const float* Wk    = (const float*)d_in[7];
  const float* bk    = (const float*)d_in[8];
  const float* Wv    = (const float*)d_in[9];
  const float* bv    = (const float*)d_in[10];
  const float* W00   = (const float*)d_in[11];
  const float* b00   = (const float*)d_in[12];
  const float* W1    = (const float*)d_in[13];
  const float* b1    = (const float*)d_in[14];
  const float* W2    = (const float*)d_in[15];
  const float* b2    = (const float*)d_in[16];
  const float* ls    = (const float*)d_in[17];

  char* ws = (char*)d_ws;
  float*    attnWS = (float*)(ws + WS_ATTN);
  _Float16* attnF  = (_Float16*)(ws + WS_ATTNF);
  _Float16* W00F   = (_Float16*)(ws + WS_W00F);
  _Float16* W1F    = (_Float16*)(ws + WS_W1F);
  _Float16* WvF    = (_Float16*)(ws + WS_WVF);
  _Float16* WkF    = (_Float16*)(ws + WS_WKF);

  prep_kernel<<<256, 256, 0, stream>>>(Wk, Wv, W00, W1, attnWS, W00F, W1F, WvF, WkF);

  attn_accum_kernel<<<dim3(K1_GRIDX, BATCH * 4), 128, 0, stream>>>(
      feat, coord, Wq, bq, bk, ls, WkF, attnWS);

  softmax_kernel<<<BATCH * 4, 64, 0, stream>>>(attnWS, attnF);

  fused_kernel<<<(BATCH * NPOS) / 32, 64, 0, stream>>>(
      feat, inp, coord, cell, scale, bv, b00, b1, W2, b2, ls,
      attnF, W00F, W1F, WvF, (float*)d_out);
}
